// Attention_37598143709539
// MI455X (gfx1250) — matrix operand order repaired and checked by reading, not compiled, not run on hardware
//
#include <hip/hip_runtime.h>
#include <hip/hip_bf16.h>
#include <stddef.h>
#include <stdint.h>

// Problem constants (from reference)
#define BATCH  4
#define SEQ    2048
#define DMODEL 512
#define HEADS  8
#define DIMH   64
#define INNER  512          // HEADS*DIMH
#define QKVC   1536         // 3*INNER
#define LOG2E  1.44269504f
#define SCALE  0.125f       // 64^-0.5

typedef __bf16 bf16_t;
typedef __bf16 v16bf __attribute__((ext_vector_type(16)));
typedef __bf16 v8bf  __attribute__((ext_vector_type(8)));
typedef float  v8f   __attribute__((ext_vector_type(8)));
typedef unsigned int u32;
typedef u32 v4u __attribute__((ext_vector_type(4)));
typedef int  v8i __attribute__((ext_vector_type(8)));
typedef int  v4i __attribute__((ext_vector_type(4)));

// TDM availability (6-arg amdgpu-toolchain form, per compile probe).
#if defined(__HIP_DEVICE_COMPILE__) && defined(__has_builtin)
#  if __has_builtin(__builtin_amdgcn_tensor_load_to_lds) && \
      __has_builtin(__builtin_amdgcn_s_wait_tensorcnt)
#    define USE_TDM 1
#  endif
#endif
#ifndef USE_TDM
#  define USE_TDM 0
#endif

// D = A(16x32 bf16) * B(32x16 bf16) + C(16x16 f32)
__device__ __forceinline__ v8f wmma_bf16(v16bf a, v16bf b, v8f c) {
    return __builtin_amdgcn_wmma_f32_16x16x32_bf16(
        /*neg_a=*/false, a, /*neg_b=*/false, b,
        /*c_mod=*/(short)0, c, /*reuse_a=*/false, /*reuse_b=*/false);
}

#if USE_TDM
// TDM descriptor (ISA ch.8): load a 16x32 fp32 tile with row stride 2048
// elements from global into LDS. Groups 2/3 zero (2-D tile).
__device__ __forceinline__ void tdm_load_tile(const float* gsrc, void* lds_dst) {
    const unsigned long long ga = (unsigned long long)(uintptr_t)gsrc;
    v4u g0;
    g0[0] = 1u;                                   // count=1, user descriptor
    g0[1] = (u32)(uintptr_t)lds_dst;              // lds_addr (byte offset)
    g0[2] = (u32)ga;                              // global_addr[31:0]
    g0[3] = (u32)((ga >> 32) & 0x01FFFFFFu) | 0x80000000u; // addr[56:32]|type=2
    v8i g1;
    g1[0] = (int)(2u << 16);      // wg_mask=0, data_size=2 (4 bytes)
    g1[1] = (int)(2048u << 16);   // tensor_dim0[15:0] = 2048 (bits 63:48)
    g1[2] = (int)(2048u << 16);   // tensor_dim0 hi=0 | tensor_dim1[15:0]=2048
    g1[3] = (int)(32u << 16);     // tensor_dim1 hi=0 | tile_dim0=32
    g1[4] = 16;                   // tile_dim1=16, tile_dim2=0
    g1[5] = 2048;                 // tensor_dim0_stride[31:0] = 2048
    g1[6] = 0;                    // stride hi | tensor_dim1_stride lo
    g1[7] = 0;
    const v4i gz4 = {0, 0, 0, 0};
    const v8i gz8 = {0, 0, 0, 0, 0, 0, 0, 0};
    __builtin_amdgcn_tensor_load_to_lds(g0, g1, gz4, gz4, gz8, 0);
}
#endif

// ---------------------------------------------------------------------------
// Kernel 1: qkv = x[8192x512] @ Wqkv[512x1536]  (fp32 in, bf16 out, scattered
// into Q[b,h,n,d], K[b,h,n,d], Vt[b,h,d,n]).  One wave per 16x16 tile.
// ---------------------------------------------------------------------------
__global__ __launch_bounds__(256) void qkv_proj_kernel(
    const float* __restrict__ x, const float* __restrict__ Wqkv,
    bf16_t* __restrict__ Q, bf16_t* __restrict__ K, bf16_t* __restrict__ Vt)
{
    const int lane  = threadIdx.x & 31;
    const int wave  = threadIdx.x >> 5;
    const int tile  = blockIdx.x * 8 + wave;        // 512*96 tiles total
    const int NT    = QKVC / 16;                    // 96
    const int mt    = tile / NT;
    const int nt    = tile % NT;
    const int r     = lane & 15;
    const int hi    = lane >> 4;
    const int base1 = 8 * hi;
    const int row   = mt * 16 + r;                  // [0, 8192)
    const int col   = nt * 16 + r;                  // this lane's B column

    v8f acc = {};
    for (int k0 = 0; k0 < DMODEL; k0 += 32) {
        // A fragment: two contiguous 8-float runs of x row, cast to bf16
        const float* xa = x + (size_t)row * DMODEL + k0 + 16 * hi;
        v16bf a;
        #pragma unroll
        for (int i = 0; i < 8; ++i) {
            a[i]     = (bf16_t)xa[i];
            a[8 + i] = (bf16_t)xa[8 + i];
        }
        // B fragment: 16 consecutive K-rows of Wqkv at this lane's column
        const float* wb = Wqkv + (size_t)(k0 + 16 * hi) * QKVC + col;
        v16bf b;
        #pragma unroll
        for (int i = 0; i < 16; ++i) b[i] = (bf16_t)wb[(size_t)i * QKVC];
        acc = wmma_bf16(a, b, acc);
    }

    // Scatter: C element e -> row mt*16 + e + 8*hi, column col.
    #pragma unroll
    for (int e = 0; e < 8; ++e) {
        const int mrow = mt * 16 + e + 8 * hi;      // 2048 % 16 == 0: no b-crossing
        const int bb = mrow >> 11, n = mrow & 2047;
        const bf16_t val = (bf16_t)acc[e];
        if (col < INNER) {
            const int h = col >> 6, d = col & 63;
            Q[(((size_t)(bb * HEADS + h) * SEQ) + n) * DIMH + d] = val;
        } else if (col < 2 * INNER) {
            const int cc = col - INNER;
            const int h = cc >> 6, d = cc & 63;
            K[(((size_t)(bb * HEADS + h) * SEQ) + n) * DIMH + d] = val;
        } else {
            const int cc = col - 2 * INNER;
            const int h = cc >> 6, d = cc & 63;
            Vt[(((size_t)(bb * HEADS + h) * DIMH) + d) * SEQ + n] = val;  // transposed
        }
    }
}

// ---------------------------------------------------------------------------
// Kernel 2: fused flash attention with conv-bias.  One block = (b, 16-query
// tile); wave w = head w.  cd tile streamed via TDM into double-buffered LDS
// (wave 0 drives the DMA), shared by all 8 heads.
// ---------------------------------------------------------------------------
__global__ __launch_bounds__(256) void attn_kernel(
    const bf16_t* __restrict__ Q, const bf16_t* __restrict__ K,
    const bf16_t* __restrict__ Vt, const float* __restrict__ cd,
    const float* __restrict__ rel_w, const float* __restrict__ rel_b,
    bf16_t* __restrict__ O)
{
    __shared__ float  lds_cd[2][3][16][32];  // 12 KB: double-buffered bias tiles
    __shared__ bf16_t lds_P[HEADS][16][32];  //  8 KB: per-wave P re-layout buffer

    const int tid  = threadIdx.x;
    const int lane = tid & 31;
    const int h    = tid >> 5;                  // head == wave
    const int b    = blockIdx.x >> 7;           // 128 i-tiles per batch
    const int i0   = (blockIdx.x & 127) * 16;
    const int r    = lane & 15;
    const int hi   = lane >> 4;
    const int base1 = 8 * hi;

    const float w0 = rel_w[h * 3 + 0], w1 = rel_w[h * 3 + 1];
    const float w2 = rel_w[h * 3 + 2], rb = rel_b[h];

    // Q A-fragments (dims 0..31 and 32..63), resident for the whole j loop.
    v16bf aq[2];
    {
        const bf16_t* qrow = Q + (((size_t)(b * HEADS + h) * SEQ) + i0 + r) * DIMH;
        #pragma unroll
        for (int t = 0; t < 2; ++t) {
            v8bf lo = *(const v8bf*)(qrow + 32 * t + 16 * hi);
            v8bf hh = *(const v8bf*)(qrow + 32 * t + 16 * hi + 8);
            #pragma unroll
            for (int i = 0; i < 8; ++i) { aq[t][i] = lo[i]; aq[t][8 + i] = hh[i]; }
        }
    }

    float m[8], l[8];
    #pragma unroll
    for (int e = 0; e < 8; ++e) { m[e] = -1e30f; l[e] = 0.f; }
    v8f o0 = {}, o1 = {}, o2 = {}, o3 = {};

    const bf16_t* Kb  = K  + ((size_t)(b * HEADS + h) * SEQ) * DIMH;
    const bf16_t* Vb  = Vt + ((size_t)(b * HEADS + h) * DIMH) * SEQ;
    const float*  cdb = cd + (size_t)b * 3 * SEQ * SEQ + (size_t)i0 * SEQ;

    int cur = 0;
#if USE_TDM
    if (h == 0) {            // wave 0 drives the tensor DMA (EXEC-independent)
        #pragma unroll
        for (int c = 0; c < 3; ++c)
            tdm_load_tile(cdb + (size_t)c * SEQ * SEQ, &lds_cd[0][c][0][0]);
    }
#endif

    #pragma unroll 1
    for (int j0 = 0; j0 < SEQ; j0 += 32) {
#if USE_TDM
        if (h == 0) {
            if (j0 + 32 < SEQ) {
                // Prefetch next tile into the other buffer, then wait until
                // only those 3 are outstanding => current tile is resident.
                #pragma unroll
                for (int c = 0; c < 3; ++c)
                    tdm_load_tile(cdb + (size_t)c * SEQ * SEQ + (j0 + 32),
                                  &lds_cd[cur ^ 1][c][0][0]);
                __builtin_amdgcn_s_wait_tensorcnt(3);
            } else {
                __builtin_amdgcn_s_wait_tensorcnt(0);
            }
        }
        __syncthreads();                      // current buffer visible to all
#else
        __syncthreads();                      // previous tile fully consumed
        // Cooperative cd tile load: 3*16*32 floats, 6 per thread.
        #pragma unroll
        for (int q = 0; q < 6; ++q) {
            const int e  = tid + 256 * q;
            const int c  = e >> 9;
            const int rm = e & 511;
            const int ri = rm >> 5, ci = rm & 31;
            lds_cd[0][c][ri][ci] =
                cdb[(size_t)c * SEQ * SEQ + (size_t)ri * SEQ + j0 + ci];
        }
        __syncthreads();
#endif

        // S = Q K^T  (two 16x16 column tiles, K-dim 64 in two steps)
        v8f s0 = {}, s1 = {};
        #pragma unroll
        for (int t = 0; t < 2; ++t) {
            const bf16_t* k0p = Kb + (size_t)(j0 + r)      * DIMH + 32 * t + 16 * hi;
            const bf16_t* k1p = Kb + (size_t)(j0 + 16 + r) * DIMH + 32 * t + 16 * hi;
            const v16bf bk0 = *(const v16bf*)k0p;   // 16 contiguous bf16, 32B aligned
            const v16bf bk1 = *(const v16bf*)k1p;
            s0 = wmma_bf16(aq[t], bk0, s0);
            s1 = wmma_bf16(aq[t], bk1, s1);
        }

        // Online softmax: scale + conv bias, row max/sum via 16-lane shuffles.
        float alpha[8];
        #pragma unroll
        for (int e = 0; e < 8; ++e) {
            const int M = e + 8 * hi;
            const float bias0 = w0 * lds_cd[cur][0][M][r]
                              + w1 * lds_cd[cur][1][M][r]
                              + w2 * lds_cd[cur][2][M][r] + rb;
            const float bias1 = w0 * lds_cd[cur][0][M][r + 16]
                              + w1 * lds_cd[cur][1][M][r + 16]
                              + w2 * lds_cd[cur][2][M][r + 16] + rb;
            const float f0 = s0[e] * SCALE + bias0;
            const float f1 = s1[e] * SCALE + bias1;
            float t = fmaxf(f0, f1);
            t = fmaxf(t, __shfl_xor(t, 1, 32));
            t = fmaxf(t, __shfl_xor(t, 2, 32));
            t = fmaxf(t, __shfl_xor(t, 4, 32));
            t = fmaxf(t, __shfl_xor(t, 8, 32));
            const float mnew = fmaxf(m[e], t);
            alpha[e] = exp2f((m[e] - mnew) * LOG2E);
            const float p0 = exp2f((f0 - mnew) * LOG2E);
            const float p1 = exp2f((f1 - mnew) * LOG2E);
            float rs = p0 + p1;
            rs += __shfl_xor(rs, 1, 32);
            rs += __shfl_xor(rs, 2, 32);
            rs += __shfl_xor(rs, 4, 32);
            rs += __shfl_xor(rs, 8, 32);
            l[e] = l[e] * alpha[e] + rs;
            m[e] = mnew;
            lds_P[h][M][r]      = (bf16_t)p0;   // wave-private region: DS in-order,
            lds_P[h][M][r + 16] = (bf16_t)p1;   // no barrier needed
        }
        #pragma unroll
        for (int e = 0; e < 8; ++e) {
            o0[e] *= alpha[e]; o1[e] *= alpha[e];
            o2[e] *= alpha[e]; o3[e] *= alpha[e];
        }

        // Re-load P in A-fragment layout from LDS.
        v16bf ap;
        {
            const bf16_t* prow = &lds_P[h][r][0];
            v8bf lo = *(const v8bf*)(prow + 16 * hi);
            v8bf hh = *(const v8bf*)(prow + 16 * hi + 8);
            #pragma unroll
            for (int i = 0; i < 8; ++i) { ap[i] = lo[i]; ap[8 + i] = hh[i]; }
        }

        // O += P * V  (V transposed -> contiguous B fragments), 4 dim chunks.
        {
            const bf16_t* v0p = Vb + (size_t)(r)      * SEQ + j0 + 16 * hi;
            const bf16_t* v1p = Vb + (size_t)(16 + r) * SEQ + j0 + 16 * hi;
            const bf16_t* v2p = Vb + (size_t)(32 + r) * SEQ + j0 + 16 * hi;
            const bf16_t* v3p = Vb + (size_t)(48 + r) * SEQ + j0 + 16 * hi;
            o0 = wmma_bf16(ap, *(const v16bf*)v0p, o0);
            o1 = wmma_bf16(ap, *(const v16bf*)v1p, o1);
            o2 = wmma_bf16(ap, *(const v16bf*)v2p, o2);
            o3 = wmma_bf16(ap, *(const v16bf*)v3p, o3);
        }

#if USE_TDM
        __syncthreads();     // all reads of lds_cd[cur] done before DMA reuses it
        cur ^= 1;
#endif
    }

    // Epilogue: divide by row sums, store bf16 as O[b, n, h*64 + d].
    #pragma unroll
    for (int e = 0; e < 8; ++e) {
        const int M = e + 8 * hi;
        const float inv = 1.0f / l[e];
        bf16_t* od = O + ((size_t)(b * SEQ + i0 + M)) * INNER + h * DIMH;
        od[r]      = (bf16_t)(o0[e] * inv);
        od[16 + r] = (bf16_t)(o1[e] * inv);
        od[32 + r] = (bf16_t)(o2[e] * inv);
        od[48 + r] = (bf16_t)(o3[e] * inv);
    }
}

// ---------------------------------------------------------------------------
// Kernel 3: out = O[8192x512](bf16) @ Wout[512x512] + bout, fp32 result.
// ---------------------------------------------------------------------------
__global__ __launch_bounds__(256) void out_proj_kernel(
    const bf16_t* __restrict__ Obuf, const float* __restrict__ Wout,
    const float* __restrict__ bout, float* __restrict__ out)
{
    const int lane  = threadIdx.x & 31;
    const int wave  = threadIdx.x >> 5;
    const int tile  = blockIdx.x * 8 + wave;      // 512*32 tiles
    const int NT    = DMODEL / 16;                // 32
    const int mt    = tile / NT;
    const int nt    = tile % NT;
    const int r     = lane & 15;
    const int hi    = lane >> 4;
    const int base1 = 8 * hi;
    const int row   = mt * 16 + r;
    const int col   = nt * 16 + r;

    v8f acc = {};
    for (int k0 = 0; k0 < INNER; k0 += 32) {
        const bf16_t* ar = Obuf + (size_t)row * INNER + k0 + 16 * hi;
        v8bf lo = *(const v8bf*)ar;
        v8bf hh = *(const v8bf*)(ar + 8);
        v16bf a;
        #pragma unroll
        for (int i = 0; i < 8; ++i) { a[i] = lo[i]; a[8 + i] = hh[i]; }
        const float* wb = Wout + (size_t)(k0 + 16 * hi) * DMODEL + col;
        v16bf bfr;
        #pragma unroll
        for (int i = 0; i < 16; ++i) bfr[i] = (bf16_t)wb[(size_t)i * DMODEL];
        acc = wmma_bf16(a, bfr, acc);
    }

    const float bias = bout[col];
    #pragma unroll
    for (int e = 0; e < 8; ++e) {
        const int mrow = mt * 16 + e + 8 * hi;
        out[(size_t)mrow * DMODEL + col] = acc[e] + bias;
    }
}

// ---------------------------------------------------------------------------
extern "C" void kernel_launch(void* const* d_in, const int* in_sizes, int n_in,
                              void* d_out, int out_size, void* d_ws, size_t ws_size,
                              hipStream_t stream)
{
    const float* x    = (const float*)d_in[0];
    const float* cd   = (const float*)d_in[1];
    const float* Wqkv = (const float*)d_in[2];
    const float* Wout = (const float*)d_in[3];
    const float* bout = (const float*)d_in[4];
    const float* relw = (const float*)d_in[5];
    const float* relb = (const float*)d_in[6];
    float* out = (float*)d_out;

    // Workspace: Q,K (row-major [b,h,n,d]), Vt ([b,h,d,n]), O ([b,n,512]),
    // all bf16: 4 x 8 MB = 32 MB.
    const size_t elems = (size_t)BATCH * HEADS * SEQ * DIMH;  // 4,194,304
    bf16_t* Q  = (bf16_t*)d_ws;
    bf16_t* K  = Q  + elems;
    bf16_t* Vt = K  + elems;
    bf16_t* O  = Vt + elems;

    // 1) QKV projection: (8192/16)*(1536/16) = 49152 tiles, 8 waves/block.
    qkv_proj_kernel<<<dim3(6144), dim3(256), 0, stream>>>(x, Wqkv, Q, K, Vt);
    // 2) Fused attention: B * N/16 = 512 blocks, one head per wave.
    attn_kernel<<<dim3(512), dim3(256), 0, stream>>>(Q, K, Vt, cd, relw, relb, O);
    // 3) Output projection: (8192/16)*(512/16) = 16384 tiles, 8 waves/block.
    out_proj_kernel<<<dim3(2048), dim3(256), 0, stream>>>(O, Wout, bout, out);
}
